// Musicmamba_28870770164060
// MI455X (gfx1250) — matrix operand order repaired and checked by reading, not compiled, not run on hardware
//
#include <hip/hip_runtime.h>
#include <hip/hip_bf16.h>
#include <math.h>

// Model dims (fixed by the reference)
#define VOCAB 2048
#define DMODEL 128
#define NLAYER 32
#define DI 256           // d_inner (I)
#define NS 16            // d_state (N)
#define KC 4             // conv kernel
#define RR 8             // dt rank
#define BB 8
#define LL 2048
#define BLROWS (BB*LL)   // 16384
#define H2 (2*DI)        // 512
#define XPD (RR + 2*NS)  // 40
#define MAXK 256         // largest GEMM K (out_proj)

typedef __attribute__((ext_vector_type(16))) __bf16 v16bf;
typedef __attribute__((ext_vector_type(8)))  __bf16 v8bf;
typedef __attribute__((ext_vector_type(8)))  float  v8f;

__device__ __forceinline__ float silu_f(float x) {
    return x * (1.0f / (1.0f + __expf(-x)));
}

// ---------------------------------------------------------------------------
// Embedding gather: h[row, c] = embed[x[row], c]
// ---------------------------------------------------------------------------
__global__ void embed_kernel(const int* __restrict__ x,
                             const float* __restrict__ embed,
                             float* __restrict__ h) {
    int idx = blockIdx.x * blockDim.x + threadIdx.x;
    if (idx >= BLROWS * DMODEL) return;
    int row = idx / DMODEL;
    int c   = idx - row * DMODEL;
    h[idx] = embed[(size_t)x[row] * DMODEL + c];
}

// ---------------------------------------------------------------------------
// fp32 -> bf16 bulk convert (weights, once per call)
// ---------------------------------------------------------------------------
__global__ void f32_to_bf16_kernel(const float* __restrict__ src,
                                   __bf16* __restrict__ dst, int n) {
    int idx = blockIdx.x * blockDim.x + threadIdx.x;
    if (idx >= n) return;
    dst[idx] = (__bf16)src[idx];
}

// ---------------------------------------------------------------------------
// RMSNorm over DMODEL, emit bf16 activations for WMMA. One wave per row.
// ---------------------------------------------------------------------------
__global__ void rmsnorm_bf16_kernel(const float* __restrict__ h,
                                    const float* __restrict__ w,
                                    __bf16* __restrict__ out, int rows) {
    int wave = threadIdx.x >> 5;
    int lane = threadIdx.x & 31;
    int row  = blockIdx.x * (blockDim.x >> 5) + wave;
    if (row >= rows) return;
    const float* hr = h + (size_t)row * DMODEL;
    float ss = 0.f;
    #pragma unroll
    for (int c = lane; c < DMODEL; c += 32) { float v = hr[c]; ss += v * v; }
    #pragma unroll
    for (int o = 16; o > 0; o >>= 1) ss += __shfl_xor(ss, o, 32);
    float inv = rsqrtf(ss / (float)DMODEL + 1e-5f);
    #pragma unroll
    for (int c = lane; c < DMODEL; c += 32)
        out[(size_t)row * DMODEL + c] = (__bf16)(hr[c] * inv * w[c]);
}

// ---------------------------------------------------------------------------
// WMMA bf16 GEMM: C(MxN) = A(MxK) * B(NxK)^T  (+bias[N]) (+residual[MxN])
//
// Block = 256 threads = 8 waves. grid.y = M/16: one 16-row A panel per block
// row. The panel (16 x K bf16, <= 8 KB) is staged into LDS with the CDNA5
// async copy engine (global_load_async_to_lds_b128, tracked by ASYNCcnt) —
// no VGPR round-trip — and shared by all 8 waves (8x cut in A global
// traffic). Each wave computes TWO adjacent 16x16 N tiles, reusing its A
// fragment in registers: 2 WMMAs per K-step. A/B k order contiguous (16h+s); C/D layout per CDNA5 ISA
// 7.12.2. All WMMA-adjacent control flow is wave-uniform (EXEC all-1s).
// ---------------------------------------------------------------------------
__global__ void gemm_bf16_wmma(const __bf16* __restrict__ A,
                               const __bf16* __restrict__ B,
                               float* __restrict__ C,
                               const float* __restrict__ bias,
                               const float* __restrict__ residual,
                               int N, int K) {
    __shared__ __bf16 sA[16 * MAXK];            // 8 KB

    int tileM = blockIdx.y;
    // ---- async A panel stage: 16 rows x K, contiguous in global ----------
    // LDS dest address = low 32 bits of the flat address (ISA 10.2: the
    // LDS aperture maps flat->LDS by truncating to addr[31:0]).
    {
        const __bf16* apanel = A + (size_t)tileM * 16 * K;
        int chunks = (16 * K) >> 3;             // 16-byte chunks (256 or 512)
        for (int c = threadIdx.x; c < chunks; c += blockDim.x) {
            unsigned ldsAddr = (unsigned)(size_t)&sA[c << 3];
            const __bf16* g  = apanel + (c << 3);
            asm volatile("global_load_async_to_lds_b128 %0, %1, off"
                         :: "v"(ldsAddr), "v"(g) : "memory");
        }
        asm volatile("s_wait_asynccnt 0" ::: "memory");
    }
    __syncthreads();

    int wave = threadIdx.x >> 5;
    int lane = threadIdx.x & 31;
    int jobs = N >> 5;                          // N-tile pairs
    int job  = blockIdx.x * (blockDim.x >> 5) + wave;
    if (job >= jobs) return;                    // wave-uniform

    int hi  = lane >> 4;
    int row = lane & 15;                        // A/B: M or N = lane&15
    int kA  = hi * 16;                          // A: K 0..15 / 16..31 contiguous
    int kB  = hi * 16;                          // B: K 0..15 / 16..31 contiguous

    int n0 = (job << 5) + row;                  // B row, first tile
    const __bf16* b0row = B + (size_t)n0 * K;
    const __bf16* b1row = b0row + (size_t)16 * K;

    v8f acc0 = {}, acc1 = {};
    for (int k0 = 0; k0 < K; k0 += 32) {
        if (k0 + 32 < K) {                      // -> global_prefetch_b8
            __builtin_prefetch(b0row + k0 + kB + 32, 0, 1);
            __builtin_prefetch(b1row + k0 + kB + 32, 0, 1);
        }
        union { v16bf v; v8bf h[2]; } ua;       // A fragment from LDS (ds_load_b128)
        ua.h[0] = *reinterpret_cast<const v8bf*>(&sA[row * K + k0 + kA]);
        ua.h[1] = *reinterpret_cast<const v8bf*>(&sA[row * K + k0 + kA + 8]);
        v16bf bv0 = *reinterpret_cast<const v16bf*>(b0row + k0 + kB);
        v16bf bv1 = *reinterpret_cast<const v16bf*>(b1row + k0 + kB);
        acc0 = __builtin_amdgcn_wmma_f32_16x16x32_bf16(
            false, ua.v, false, bv0, (short)0, acc0, false, false);
        acc1 = __builtin_amdgcn_wmma_f32_16x16x32_bf16(
            false, ua.v, false, bv1, (short)0, acc1, false, false);
    }

    // C/D layout: VGPR r -> M = r (lanes 0-15) / r+8 (lanes 16-31), N = lane&15
    int mbase = (tileM << 4) + (hi ? 8 : 0);
    int cn0   = (job << 5) + row;
    int cn1   = cn0 + 16;
    float bv0 = bias ? bias[cn0] : 0.0f;
    float bv1 = bias ? bias[cn1] : 0.0f;
    #pragma unroll
    for (int r = 0; r < 8; ++r) {
        size_t i0 = (size_t)(mbase + r) * N + cn0;
        size_t i1 = (size_t)(mbase + r) * N + cn1;
        float v0 = acc0[r] + bv0;
        float v1 = acc1[r] + bv1;
        if (residual) { v0 += residual[i0]; v1 += residual[i1]; }
        C[i0] = v0;
        C[i1] = v1;
    }
}

// ---------------------------------------------------------------------------
// Causal depthwise conv (K=4) over L + bias + SiLU.  Input hs = hz[:, 0:DI].
// ---------------------------------------------------------------------------
__global__ void conv_silu_kernel(const float* __restrict__ hz,
                                 const float* __restrict__ cw,   // (DI,KC)
                                 const float* __restrict__ cb,   // (DI)
                                 float* __restrict__ hs) {
    int idx = blockIdx.x * blockDim.x + threadIdx.x;
    if (idx >= BB * LL * DI) return;
    int i = idx % DI;
    int l = (idx / DI) % LL;
    int b = idx / (DI * LL);
    float acc = cb[i];
    #pragma unroll
    for (int k = 0; k < KC; ++k) {
        int lp = l - (KC - 1) + k;
        if (lp >= 0)
            acc += hz[((size_t)(b * LL + lp)) * H2 + i] * cw[i * KC + k];
    }
    hs[idx] = silu_f(acc);
}

// ---------------------------------------------------------------------------
// x_proj: xp(BL x 40) = hs(BL x 256) @ xw(40 x 256)^T  (small N -> VALU)
// ---------------------------------------------------------------------------
__global__ void xproj_kernel(const float* __restrict__ hs,
                             const float* __restrict__ xw,
                             float* __restrict__ xp) {
    int idx = blockIdx.x * blockDim.x + threadIdx.x;
    if (idx >= BLROWS * XPD) return;
    int row = idx / XPD;
    int c   = idx - row * XPD;
    const float* a = hs + (size_t)row * DI;
    const float* w = xw + (size_t)c * DI;
    float acc = 0.f;
    #pragma unroll 8
    for (int k = 0; k < DI; ++k) acc += a[k] * w[k];
    xp[idx] = acc;
}

// ---------------------------------------------------------------------------
// dt_proj (rank 8) + softplus: dtp(BL x 256)
// ---------------------------------------------------------------------------
__global__ void dtproj_softplus_kernel(const float* __restrict__ xp,
                                       const float* __restrict__ dw,  // (DI,RR)
                                       const float* __restrict__ db,  // (DI)
                                       float* __restrict__ dtp) {
    int idx = blockIdx.x * blockDim.x + threadIdx.x;
    if (idx >= BLROWS * DI) return;
    int row = idx / DI;
    int i   = idx - row * DI;
    const float* d = xp + (size_t)row * XPD;   // first RR entries = dt_raw
    float acc = db[i];
    #pragma unroll
    for (int r = 0; r < RR; ++r) acc += d[r] * dw[i * RR + r];
    dtp[idx] = (acc > 20.f) ? acc : log1pf(__expf(acc));
}

// ---------------------------------------------------------------------------
// Selective scan: one lane per (b,i,n); recurrence over L in registers;
// y[b,l,i] = sum_n(h_n * C_n) + u*D.  16-lane shuffle reduction per step.
// ---------------------------------------------------------------------------
__global__ void scan_kernel(const float* __restrict__ dtp,
                            const float* __restrict__ hs,
                            const float* __restrict__ xp,
                            const float* __restrict__ A_log,  // (DI,NS)
                            const float* __restrict__ Dv,     // (DI)
                            float* __restrict__ y) {
    int tid = blockIdx.x * blockDim.x + threadIdx.x;
    if (tid >= BB * DI * NS) return;
    int n  = tid & (NS - 1);
    int bi = tid >> 4;
    int i  = bi % DI;
    int b  = bi / DI;
    float An = -__expf(A_log[i * NS + n]);
    float Di = Dv[i];
    float st = 0.f;
    for (int l = 0; l < LL; ++l) {
        size_t r = (size_t)b * LL + l;
        float dt = dtp[r * DI + i];
        float u  = hs [r * DI + i];
        const float* xr = xp + r * XPD;
        float Bn = xr[RR + n];
        float Cn = xr[RR + NS + n];
        st = __expf(dt * An) * st + dt * Bn * u;
        float part = st * Cn;
        #pragma unroll
        for (int o = 1; o < NS; o <<= 1) part += __shfl_xor(part, o, 32);
        if (n == 0) y[r * DI + i] = part + u * Di;
    }
}

// ---------------------------------------------------------------------------
// y * silu(gate) -> bf16 for out_proj WMMA.  gate = hz[:, DI:2*DI]
// ---------------------------------------------------------------------------
__global__ void gate_mul_bf16_kernel(const float* __restrict__ y,
                                     const float* __restrict__ hz,
                                     __bf16* __restrict__ ybf) {
    int idx = blockIdx.x * blockDim.x + threadIdx.x;
    if (idx >= BLROWS * DI) return;
    int row = idx / DI;
    int i   = idx - row * DI;
    float g = hz[(size_t)row * H2 + DI + i];
    ybf[idx] = (__bf16)(y[idx] * silu_f(g));
}

// ---------------------------------------------------------------------------
// Host side
// ---------------------------------------------------------------------------
static void launch_gemm(const __bf16* A, const __bf16* B, float* C,
                        const float* bias, const float* res,
                        int M, int N, int K, hipStream_t s) {
    int jobs = N / 32;                          // N-tile pairs per M panel
    const int wpb = 8;                          // 8 waves = 256 threads / block
    dim3 grid((jobs + wpb - 1) / wpb, M / 16);
    gemm_bf16_wmma<<<grid, wpb * 32, 0, s>>>(A, B, C, bias, res, N, K);
}

extern "C" void kernel_launch(void* const* d_in, const int* in_sizes, int n_in,
                              void* d_out, int out_size, void* d_ws, size_t ws_size,
                              hipStream_t stream) {
    const int*   x          = (const int*)  d_in[0];
    const float* embed      = (const float*)d_in[1];
    const float* in_proj_w  = (const float*)d_in[2];   // (NL, 512, 128)
    const float* conv_w     = (const float*)d_in[3];   // (NL, 256, 4)
    const float* conv_b     = (const float*)d_in[4];   // (NL, 256)
    const float* x_proj_w   = (const float*)d_in[5];   // (NL, 40, 256)
    const float* dt_proj_w  = (const float*)d_in[6];   // (NL, 256, 8)
    const float* dt_proj_b  = (const float*)d_in[7];   // (NL, 256)
    const float* A_log      = (const float*)d_in[8];   // (NL, 256, 16)
    const float* Dp         = (const float*)d_in[9];   // (NL, 256)
    const float* out_proj_w = (const float*)d_in[10];  // (NL, 128, 256)
    const float* norm_w     = (const float*)d_in[11];  // (NL, 128)
    const float* norm_f_w   = (const float*)d_in[12];  // (128)
    const float* head_w     = (const float*)d_in[13];  // (2048, 128)
    const float* head_b     = (const float*)d_in[14];  // (2048)
    float* out = (float*)d_out;

    // workspace carve-up (~115 MB total)
    char* ws = (char*)d_ws;
    size_t off = 0;
    auto carve = [&](size_t bytes) -> char* {
        char* p = ws + off;
        off = (off + bytes + 255) & ~(size_t)255;
        return p;
    };
    float*  h       = (float*) carve((size_t)BLROWS * DMODEL * 4);
    __bf16* hn      = (__bf16*)carve((size_t)BLROWS * DMODEL * 2);
    float*  hz      = (float*) carve((size_t)BLROWS * H2     * 4);
    float*  hs      = (float*) carve((size_t)BLROWS * DI     * 4);
    float*  xp      = (float*) carve((size_t)BLROWS * XPD    * 4);
    float*  dtp     = (float*) carve((size_t)BLROWS * DI     * 4);
    float*  ybuf    = (float*) carve((size_t)BLROWS * DI     * 4);
    __bf16* ybf     = (__bf16*)carve((size_t)BLROWS * DI     * 2);
    __bf16* w_in    = (__bf16*)carve((size_t)NLAYER * H2 * DMODEL * 2);
    __bf16* w_out   = (__bf16*)carve((size_t)NLAYER * DMODEL * DI * 2);
    __bf16* w_head  = (__bf16*)carve((size_t)VOCAB * DMODEL * 2);
    (void)ws_size; (void)in_sizes; (void)n_in; (void)out_size;

    const int TB = 256;
    auto blocks = [](long n, int tb) { return (int)((n + tb - 1) / tb); };
    const int normGrid = (BLROWS + (TB / 32) - 1) / (TB / 32);   // 8 rows/block

    // 1) embedding
    embed_kernel<<<blocks((long)BLROWS * DMODEL, TB), TB, 0, stream>>>(x, embed, h);

    // 2) weights -> bf16 (once per call; weights live in L2 thereafter)
    {
        int n1 = NLAYER * H2 * DMODEL;
        int n2 = NLAYER * DMODEL * DI;
        int n3 = VOCAB * DMODEL;
        f32_to_bf16_kernel<<<blocks(n1, TB), TB, 0, stream>>>(in_proj_w,  w_in,   n1);
        f32_to_bf16_kernel<<<blocks(n2, TB), TB, 0, stream>>>(out_proj_w, w_out,  n2);
        f32_to_bf16_kernel<<<blocks(n3, TB), TB, 0, stream>>>(head_w,     w_head, n3);
    }

    // 3) layers
    for (int layer = 0; layer < NLAYER; ++layer) {
        const __bf16* wi = w_in  + (size_t)layer * H2 * DMODEL;
        const __bf16* wo = w_out + (size_t)layer * DMODEL * DI;

        rmsnorm_bf16_kernel<<<normGrid, TB, 0, stream>>>(
            h, norm_w + (size_t)layer * DMODEL, hn, BLROWS);

        // hz = hn @ in_proj_w^T   (16384 x 512, K=128)
        launch_gemm(hn, wi, hz, nullptr, nullptr, BLROWS, H2, DMODEL, stream);

        conv_silu_kernel<<<blocks((long)BLROWS * DI, TB), TB, 0, stream>>>(
            hz, conv_w + (size_t)layer * DI * KC, conv_b + (size_t)layer * DI, hs);

        xproj_kernel<<<blocks((long)BLROWS * XPD, TB), TB, 0, stream>>>(
            hs, x_proj_w + (size_t)layer * XPD * DI, xp);

        dtproj_softplus_kernel<<<blocks((long)BLROWS * DI, TB), TB, 0, stream>>>(
            xp, dt_proj_w + (size_t)layer * DI * RR, dt_proj_b + (size_t)layer * DI, dtp);

        scan_kernel<<<blocks((long)BB * DI * NS, TB), TB, 0, stream>>>(
            dtp, hs, xp, A_log + (size_t)layer * DI * NS, Dp + (size_t)layer * DI, ybuf);

        gate_mul_bf16_kernel<<<blocks((long)BLROWS * DI, TB), TB, 0, stream>>>(ybuf, hz, ybf);

        // h = h + ybf @ out_proj_w^T  (residual fused into GEMM epilogue)
        launch_gemm(ybf, wo, h, nullptr, h, BLROWS, DMODEL, DI, stream);
    }

    // 4) final norm + head
    rmsnorm_bf16_kernel<<<normGrid, TB, 0, stream>>>(h, norm_f_w, hn, BLROWS);
    // logits = hn @ head_w^T + head_b   (16384 x 2048, K=128)
    launch_gemm(hn, w_head, out, head_b, nullptr, BLROWS, VOCAB, DMODEL, stream);
}
